// GCN_57294863729308
// MI455X (gfx1250) — matrix operand order repaired and checked by reading, not compiled, not run on hardware
//
#include <hip/hip_runtime.h>

#define N_NODES  100000
#define N_EDGES  1600000
#define D_H      128
#define N_CLASSES 40
#define BN_EPS   1e-5f
#define STRIPS   ((N_NODES + 15) / 16)   // 6250 sixteen-row strips

typedef __attribute__((ext_vector_type(16))) __bf16 v16bf;
typedef __attribute__((ext_vector_type(8)))  float  v8f;

union BF4 { __bf16 h[4]; uint2 u; };

// ---------------- degree / norms ----------------
__global__ void gcn_deg_count(const int* __restrict__ idx, float* __restrict__ deg, int n) {
  int i = blockIdx.x * blockDim.x + threadIdx.x;
  if (i < n) unsafeAtomicAdd(&deg[idx[i]], 1.0f);
}

__global__ void gcn_rsqrt_clip(float* __restrict__ d, int n) {
  int i = blockIdx.x * blockDim.x + threadIdx.x;
  if (i < n) {
    float v = d[i];
    d[i] = rsqrtf(v < 1.0f ? 1.0f : v);
  }
}

// ---------------- weight packing into WMMA B-fragment layout ----------------
// Bp[(kt*8+nt)*32*16 + lane*16 + e] = bf16( W[k][n] )
//   n = nt*16 + (lane&15),  k = kt*32 + 16*(lane>>4) + e
__global__ void gcn_pack_w(const float* __restrict__ W, __bf16* __restrict__ Bp) {
  int idx = blockIdx.x * blockDim.x + threadIdx.x;   // 4*8*32*16 = 16384
  if (idx >= 16384) return;
  int e    = idx & 15;
  int lane = (idx >> 4) & 31;
  int t    = idx >> 9;            // kt*8 + nt
  int nt = t & 7, kt = t >> 3;
  int n = nt * 16 + (lane & 15);
  int k = kt * 32 + ((lane >> 4) << 4) + e;
  Bp[idx] = (__bf16)W[k * D_H + n];
}

// ---------------- elementwise: A = bf16(H * norm_src), x4 vectorized ----------------
__global__ void gcn_scale_bf16(const float* __restrict__ H, const float* __restrict__ ns,
                               __bf16* __restrict__ A, int total4) {
  int i = blockIdx.x * blockDim.x + threadIdx.x;
  if (i >= total4) return;
  float4 hv = ((const float4*)H)[i];
  float s = ns[i >> 5];                    // 32 float4 chunks per node row
  BF4 o;
  o.h[0] = (__bf16)(hv.x * s); o.h[1] = (__bf16)(hv.y * s);
  o.h[2] = (__bf16)(hv.z * s); o.h[3] = (__bf16)(hv.w * s);
  ((uint2*)A)[i] = o.u;
}

// ---------------- WMMA GEMM: C[100000,128] = A @ W(128x128) ----------------
// Persistent blocks; weight fragments staged once per block in LDS, then hoisted
// by the compiler into extended VGPRs (gfx1250 1024-VGPR file). Loop-carried
// pointers: inner loop is pure b128 A-loads + 32 WMMAs + imm-offset stores.
__global__ void __launch_bounds__(256)
gcn_gemm_wmma(const __bf16* __restrict__ A, const __bf16* __restrict__ Bp,
              float* __restrict__ C, int nstrips) {
  __shared__ uint4 Blds4[2048];                       // 32 KB: 32 frags x 32 lanes x 32B
  for (int i = threadIdx.x; i < 2048; i += 256) Blds4[i] = ((const uint4*)Bp)[i];
  __syncthreads();
  const __bf16* Blds = (const __bf16*)Blds4;

  const int waveid = threadIdx.x >> 5;
  const int lane = threadIdx.x & 31;
  const int m = lane & 15, half = lane >> 4;
  const int wstride = gridDim.x * 8;
  const int strip0 = blockIdx.x * 8 + waveid;

  // loop-carried pointers: constant stride per grid-stride iteration
  const size_t stepA = (size_t)wstride * 16 * D_H;    // bf16 elements
  const size_t stepC = (size_t)wstride * 16 * D_H;    // floats
  const __bf16* arow = A + (size_t)(strip0 * 16 + m) * D_H;
  float* crow = C + (size_t)(strip0 * 16 + 8 * half) * D_H + m;

  union Frag { v16bf v; uint4 q[2]; };

  for (int strip = strip0; strip < nstrips; strip += wstride, arow += stepA, crow += stepC) {
    v8f acc[8] = {};
#pragma unroll
    for (int kt = 0; kt < 4; ++kt) {                  // K steps of 32
      Frag a;                                         // 16-bit A 16x32 layout:
      a.q[0] = *(const uint4*)(arow + kt * 32 + half * 16);        // two contiguous
      a.q[1] = *(const uint4*)(arow + kt * 32 + half * 16 + 8);   // 8-elem runs
#pragma unroll
      for (int nt = 0; nt < 8; ++nt) {
        Frag b;
        const uint4* bptr = (const uint4*)(Blds + (((kt * 8 + nt) * 32) + lane) * 16);
        b.q[0] = bptr[0];
        b.q[1] = bptr[1];
        acc[nt] = __builtin_amdgcn_wmma_f32_16x16x32_bf16(
            false, a.v, false, b.v, (short)0, acc[nt], false, false);
      }
    }
    // C/D layout: lane = N%16 + 16*half, VGPR r holds M = r + 8*half.
    // All 64 stores are imm offsets (< 8 KB) from one per-strip base pointer.
#pragma unroll
    for (int nt = 0; nt < 8; ++nt)
#pragma unroll
      for (int r = 0; r < 8; ++r)
        crow[r * D_H + nt * 16] = acc[nt][r];
  }
}

// ---------------- SpMM: M[dst] += T[src] (one wave per edge) ----------------
__global__ void gcn_spmm(const float* __restrict__ T, const int* __restrict__ src,
                         const int* __restrict__ dst, float* __restrict__ M, int ne) {
  int w = blockIdx.x * (blockDim.x >> 5) + (threadIdx.x >> 5);
  int lane = threadIdx.x & 31;
  if (w >= ne) return;
  int s = src[w], d = dst[w];
  const float4 v = *(const float4*)(T + (size_t)s * D_H + lane * 4);
  float* o = M + (size_t)d * D_H + lane * 4;
  unsafeAtomicAdd(o + 0, v.x);
  unsafeAtomicAdd(o + 1, v.y);
  unsafeAtomicAdd(o + 2, v.z);
  unsafeAtomicAdd(o + 3, v.w);
}

// ---------------- in-place: M = M * norm_dst + b, x4 vectorized ----------------
__global__ void gcn_bias(const float* __restrict__ nd, const float* __restrict__ b,
                         float* __restrict__ M, int total4) {
  int i = blockIdx.x * blockDim.x + threadIdx.x;
  if (i >= total4) return;
  float s = nd[i >> 5];
  float4 bv = ((const float4*)b)[i & 31];
  float4 v = ((float4*)M)[i];
  v.x = v.x * s + bv.x;
  v.y = v.y * s + bv.y;
  v.z = v.z * s + bv.z;
  v.w = v.w * s + bv.w;
  ((float4*)M)[i] = v;
}

// ---------------- BatchNorm stats: per-feature sum / sumsq ----------------
__global__ void gcn_bn_stats(const float* __restrict__ H, float* __restrict__ sums,
                             float* __restrict__ sumsq, int nnodes) {
  int f = threadIdx.x;                    // blockDim.x == 128
  int start = blockIdx.x * 256;
  int end = start + 256; if (end > nnodes) end = nnodes;
  float s = 0.f, s2 = 0.f;
  for (int nd = start; nd < end; ++nd) {
    float v = H[(size_t)nd * D_H + f];
    s += v; s2 += v * v;
  }
  unsafeAtomicAdd(&sums[f], s);
  unsafeAtomicAdd(&sumsq[f], s2);
}

// ---------------- A = bf16(relu(BN(H)) * norm_src), x4 vectorized ----------------
__global__ void gcn_bn_relu_scale(const float* __restrict__ H, const float* __restrict__ sums,
                                  const float* __restrict__ sumsq, const float* __restrict__ g,
                                  const float* __restrict__ be, const float* __restrict__ ns,
                                  __bf16* __restrict__ A, int total4) {
  int i = blockIdx.x * blockDim.x + threadIdx.x;
  if (i >= total4) return;
  const float invN = 1.0f / (float)N_NODES;
  int f0 = (i & 31) * 4;
  float s = ns[i >> 5];
  float4 hv = ((const float4*)H)[i];
  float hx[4] = {hv.x, hv.y, hv.z, hv.w};
  BF4 o;
#pragma unroll
  for (int j = 0; j < 4; ++j) {
    float mu  = sums[f0 + j] * invN;
    float var = sumsq[f0 + j] * invN - mu * mu;
    float v = (hx[j] - mu) * rsqrtf(var + BN_EPS) * g[f0 + j] + be[f0 + j];
    v = v > 0.f ? v : 0.f;
    o.h[j] = (__bf16)(v * s);
  }
  ((uint2*)A)[i] = o.u;
}

// ---------------- predict + log_softmax (one thread per node) ----------------
__global__ void gcn_predict(const float* __restrict__ H, const float* __restrict__ Wp,
                            const float* __restrict__ bp, float* __restrict__ out, int nnodes) {
  int node = blockIdx.x * blockDim.x + threadIdx.x;
  if (node >= nnodes) return;
  float acc[N_CLASSES];
#pragma unroll
  for (int c = 0; c < N_CLASSES; ++c) acc[c] = bp[c];
  const float* h = H + (size_t)node * D_H;
  for (int k = 0; k < D_H; ++k) {
    float hv = h[k];
    const float* wr = Wp + k * N_CLASSES;   // uniform across lanes -> scalar loads
#pragma unroll
    for (int c = 0; c < N_CLASSES; ++c) acc[c] = fmaf(hv, wr[c], acc[c]);
  }
  float mx = acc[0];
#pragma unroll
  for (int c = 1; c < N_CLASSES; ++c) mx = fmaxf(mx, acc[c]);
  float ssum = 0.f;
#pragma unroll
  for (int c = 0; c < N_CLASSES; ++c) ssum += __expf(acc[c] - mx);
  float lse = mx + __logf(ssum);
  float* o = out + (size_t)node * N_CLASSES;
#pragma unroll
  for (int c = 0; c < N_CLASSES; ++c) o[c] = acc[c] - lse;
}

// ---------------- host ----------------
extern "C" void kernel_launch(void* const* d_in, const int* in_sizes, int n_in,
                              void* d_out, int out_size, void* d_ws, size_t ws_size,
                              hipStream_t stream) {
  const float* x   = (const float*)d_in[0];
  const int*   src = (const int*)d_in[1];
  const int*   dst = (const int*)d_in[2];
  const float* W0  = (const float*)d_in[3];
  const float* b0  = (const float*)d_in[4];
  const float* g0  = (const float*)d_in[5];
  const float* be0 = (const float*)d_in[6];
  const float* W1  = (const float*)d_in[7];
  const float* b1  = (const float*)d_in[8];
  const float* g1  = (const float*)d_in[9];
  const float* be1 = (const float*)d_in[10];
  const float* W2  = (const float*)d_in[11];
  const float* b2  = (const float*)d_in[12];
  const float* Wp  = (const float*)d_in[13];
  const float* bp  = (const float*)d_in[14];
  float* out = (float*)d_out;
  (void)in_sizes; (void)n_in; (void)out_size; (void)ws_size;

  char* ws = (char*)d_ws;
  size_t off = 0;
  auto alloc = [&](size_t bytes) -> void* {
    void* p = ws + off;
    off = (off + bytes + 255) & ~(size_t)255;
    return p;
  };
  const int total  = N_NODES * D_H;                   // 12.8M elements
  const int total4 = total / 4;                       // 3.2M float4/bf16x4 chunks
  float*  normsrc = (float*)alloc((size_t)N_NODES * 4);
  float*  normdst = (float*)alloc((size_t)N_NODES * 4);
  __bf16* A       = (__bf16*)alloc((size_t)total * 2);
  float*  T       = (float*)alloc((size_t)total * 4);
  float*  M       = (float*)alloc((size_t)total * 4);
  __bf16* Bp0     = (__bf16*)alloc(16384 * 2);
  __bf16* Bp1     = (__bf16*)alloc(16384 * 2);
  __bf16* Bp2     = (__bf16*)alloc(16384 * 2);
  float*  sums    = (float*)alloc(128 * 4);
  float*  sumsq   = (float*)alloc(128 * 4);

  // degree norms
  hipMemsetAsync(normsrc, 0, (size_t)N_NODES * 4, stream);
  hipMemsetAsync(normdst, 0, (size_t)N_NODES * 4, stream);
  gcn_deg_count<<<(N_EDGES + 255) / 256, 256, 0, stream>>>(src, normsrc, N_EDGES);
  gcn_deg_count<<<(N_EDGES + 255) / 256, 256, 0, stream>>>(dst, normdst, N_EDGES);
  gcn_rsqrt_clip<<<(N_NODES + 255) / 256, 256, 0, stream>>>(normsrc, N_NODES);
  gcn_rsqrt_clip<<<(N_NODES + 255) / 256, 256, 0, stream>>>(normdst, N_NODES);

  // pack weights into WMMA fragment layout
  gcn_pack_w<<<64, 256, 0, stream>>>(W0, Bp0);
  gcn_pack_w<<<64, 256, 0, stream>>>(W1, Bp1);
  gcn_pack_w<<<64, 256, 0, stream>>>(W2, Bp2);

  // A = bf16(x * norm_src)
  gcn_scale_bf16<<<(total4 + 255) / 256, 256, 0, stream>>>(x, normsrc, A, total4);

  const int gemm_blocks = 256;                        // persistent; grid-stride strips

  auto run_layer = [&](const __bf16* Bpk, const float* bias, const float* gamma,
                       const float* beta, bool bn) {
    gcn_gemm_wmma<<<gemm_blocks, 256, 0, stream>>>(A, Bpk, T, STRIPS);
    hipMemsetAsync(M, 0, (size_t)total * 4, stream);
    gcn_spmm<<<(N_EDGES + 7) / 8, 256, 0, stream>>>(T, src, dst, M, N_EDGES);
    gcn_bias<<<(total4 + 255) / 256, 256, 0, stream>>>(normdst, bias, M, total4);
    if (bn) {
      hipMemsetAsync(sums, 0, 128 * 4, stream);
      hipMemsetAsync(sumsq, 0, 128 * 4, stream);
      gcn_bn_stats<<<(N_NODES + 255) / 256, 128, 0, stream>>>(M, sums, sumsq, N_NODES);
      gcn_bn_relu_scale<<<(total4 + 255) / 256, 256, 0, stream>>>(
          M, sums, sumsq, gamma, beta, normsrc, A, total4);
    }
  };

  run_layer(Bp0, b0, g0, be0, true);
  run_layer(Bp1, b1, g1, be1, true);
  run_layer(Bp2, b2, nullptr, nullptr, false);        // final h lives in M

  gcn_predict<<<(N_NODES + 127) / 128, 128, 0, stream>>>(M, Wp, bp, out, N_NODES);
}
